// NonLocalBlock_37958920962730
// MI455X (gfx1250) — matrix operand order repaired and checked by reading, not compiled, not run on hardware
//
#include <hip/hip_runtime.h>
#include <hip/hip_bf16.h>

typedef __attribute__((ext_vector_type(16))) _Float16 v16h;
typedef __attribute__((ext_vector_type(8)))  _Float16 v8h;
typedef __attribute__((ext_vector_type(8)))  float    v8f;

#define B_     4
#define N_     4096      // H*W
#define C_     256
#define D_     128
#define MPAD   4096
#define MVALID 4095      // maxpool(2,1) -> N-1 valid keys
#define ROWS   (B_*N_)   // 16384

// padded LDS row strides (in _Float16 elements) to avoid bank conflicts
#define KLD    40        // 32 cols + 8 pad  (80B row stride)
#define VLD    136       // 128 cols + 8 pad (272B row stride)
#define PLD    40        // 32 cols + 8 pad

__device__ __forceinline__ v8f wmma16(v16h a, v16h b, v8f c) {
  return __builtin_amdgcn_wmma_f32_16x16x32_f16(false, a, false, b, (short)0, c,
                                                false, false);
}

// async 16B global -> LDS copy (per-lane addresses), tracked by ASYNCcnt
__device__ __forceinline__ void async_copy16(const void* gptr, void* lptr) {
  unsigned lds = (unsigned)(uintptr_t)lptr;   // low 32 bits = LDS offset
  asm volatile("global_load_async_to_lds_b128 %0, %1, off"
               :: "v"(lds), "v"(gptr) : "memory");
}
__device__ __forceinline__ void wait_async0() {
  asm volatile("s_wait_asynccnt 0" ::: "memory");
}

// ---- fragment loaders ----------------------------------------------------
// A-matrix 16x32 f16 layout: lanes 0-15 -> M=lane, VGPR0-3: K=0..7, VGPR4-7: K=16..23
//                            lanes16-31 -> M=lane-16, VGPR0-3: K=8..15, VGPR4-7: K=24..31
__device__ __forceinline__ v16h load_a_f32(const float* A, int lda, int r, int half, int k0) {
  const float* p0 = A + (size_t)r * lda + k0 + 8 * half;
  const float* p1 = p0 + 16;
  v16h a;
#pragma unroll
  for (int i = 0; i < 8; ++i) { a[i] = (_Float16)p0[i]; a[8 + i] = (_Float16)p1[i]; }
  return a;
}

__device__ __forceinline__ v16h load_a_f16(const _Float16* A, int lda, int r, int half, int k0) {
  const _Float16* p0 = A + (size_t)r * lda + k0 + 8 * half;
  v8h lo = *(const v8h*)p0;
  v8h hi = *(const v8h*)(p0 + 16);
  v16h a;
#pragma unroll
  for (int i = 0; i < 8; ++i) { a[i] = lo[i]; a[8 + i] = hi[i]; }
  return a;
}

// B-matrix 32x16 f16 layout: lane l -> N col l%16, elements 0-7: K=8h..8h+7, 8-15: K=16+8h..23+8h
__device__ __forceinline__ v16h load_b_f32(const float* Bm, int ldb, int lane, int k0, int n0) {
  const int hf = lane >> 4;
  const float* p = Bm + (size_t)k0 * ldb + n0 + (lane & 15);
  v16h b;
#pragma unroll
  for (int i = 0; i < 8; ++i) { b[i] = (_Float16)p[(size_t)(8 * hf + i) * ldb]; b[8 + i] = (_Float16)p[(size_t)(16 + 8 * hf + i) * ldb]; }
  return b;
}

__device__ __forceinline__ v16h load_b_f16(const _Float16* p, int ld, int hf) {
  v16h b;
#pragma unroll
  for (int i = 0; i < 8; ++i) { b[i] = p[(8 * hf + i) * ld]; b[8 + i] = p[(16 + 8 * hf + i) * ld]; }
  return b;
}

__device__ __forceinline__ float redmax16(float v) {
  v = fmaxf(v, __shfl_xor(v, 1, 32));
  v = fmaxf(v, __shfl_xor(v, 2, 32));
  v = fmaxf(v, __shfl_xor(v, 4, 32));
  v = fmaxf(v, __shfl_xor(v, 8, 32));
  return v;
}
__device__ __forceinline__ float redsum16(float v) {
  v += __shfl_xor(v, 1, 32);
  v += __shfl_xor(v, 2, 32);
  v += __shfl_xor(v, 4, 32);
  v += __shfl_xor(v, 8, 32);
  return v;
}

// ---- kernel 1: theta/phi/g projections (fp16 out) ------------------------
__global__ __launch_bounds__(128)
void proj_kernel(const float* __restrict__ x, const float* __restrict__ Wt,
                 const float* __restrict__ Wp, const float* __restrict__ Wg,
                 _Float16* __restrict__ theta, _Float16* __restrict__ phi,
                 _Float16* __restrict__ g) {
  const int lane = threadIdx.x & 31, wave = threadIdx.x >> 5;
  const int r = lane & 15, half = lane >> 4;
  const int t = blockIdx.y * 4 + wave;          // 0..23 (3 mats x 8 col tiles)
  const int mat = t >> 3, n0 = (t & 7) * 16;
  const float* Wm = (mat == 0) ? Wt : (mat == 1) ? Wp : Wg;
  _Float16* out = (mat == 0) ? theta : (mat == 1) ? phi : g;
  const int row0 = blockIdx.x * 16;
  const float* Arow = x + (size_t)row0 * C_;

  v8f acc = {};
#pragma unroll
  for (int k0 = 0; k0 < C_; k0 += 32) {
    v16h a = load_a_f32(Arow, C_, r, half, k0);
    v16h b = load_b_f32(Wm, D_, lane, k0, n0);
    acc = wmma16(a, b, acc);
  }
#pragma unroll
  for (int v = 0; v < 8; ++v) {
    int row = row0 + v + 8 * half;
    out[(size_t)row * D_ + n0 + r] = (_Float16)acc[v];
  }
}

// ---- kernel 2: maxpool(2,1) + transpose phi, pad to MPAD -----------------
__global__ void pool_kernel(const _Float16* __restrict__ phi,
                            const _Float16* __restrict__ g,
                            _Float16* __restrict__ phiT,   // [B][D][MPAD]
                            _Float16* __restrict__ gp) {   // [B][MPAD][D]
  int tid = blockIdx.x * blockDim.x + threadIdx.x;
  if (tid >= B_ * MPAD * D_) return;
  int d = tid & (D_ - 1);
  int m = (tid >> 7) & (MPAD - 1);
  int b = tid >> 19;
  float pv = 0.f, gv = 0.f;
  if (m < MVALID) {
    size_t base = ((size_t)b * N_ + m) * D_ + d;
    pv = fmaxf((float)phi[base], (float)phi[base + D_]);
    gv = fmaxf((float)g[base],   (float)g[base + D_]);
  }
  phiT[((size_t)b * D_ + d) * MPAD + m] = (_Float16)pv;
  gp[((size_t)b * MPAD + m) * D_ + d]   = (_Float16)gv;
}

// ---- kernel 3: flash attention with async double-buffered K/V in LDS -----
__global__ __launch_bounds__(128)
void attn_kernel(const _Float16* __restrict__ theta,
                 const _Float16* __restrict__ phiT,
                 const _Float16* __restrict__ gp,
                 _Float16* __restrict__ y) {
  // K tile: [128 d-rows][32 keys] (padded rows), V tile: [32 keys][128 d]
  __shared__ __align__(16) _Float16 Kbuf[2][128 * KLD];
  __shared__ __align__(16) _Float16 Vbuf[2][32 * VLD];
  __shared__ __align__(16) _Float16 pstage[4][16 * PLD];

  const int tid = threadIdx.x;
  const int lane = tid & 31, wave = tid >> 5;
  const int r = lane & 15, half = lane >> 4;
  const int b = blockIdx.x >> 6;
  const int qt = blockIdx.x & 63;
  const int row0 = qt * 64 + wave * 16;            // within batch
  const _Float16* Q  = theta + ((size_t)b * N_ + row0) * D_;
  const _Float16* KT = phiT + (size_t)b * D_ * MPAD;
  const _Float16* V  = gp   + (size_t)b * MPAD * D_;

  v16h qf[4];
#pragma unroll
  for (int dk = 0; dk < 4; ++dk) qf[dk] = load_a_f16(Q, D_, r, half, dk * 32);

  v8f O[8];
  float mi[8], li[8];
  v8f zero = {};
#pragma unroll
  for (int v = 0; v < 8; ++v) { O[v] = zero; mi[v] = -3.0e38f; li[v] = 0.f; }

  // cooperative async fill of one K/V tile pair (128 threads, 8 issues each)
  auto issue_tile = [&](int buf, int kt2) {
    // K: 512 chunks of 16B; chunk c -> d = c>>2, part = c&3
#pragma unroll
    for (int j = 0; j < 4; ++j) {
      int c = j * 128 + tid;
      int d = c >> 2, part = c & 3;
      async_copy16(KT + (size_t)d * MPAD + kt2 + part * 8,
                   &Kbuf[buf][d * KLD + part * 8]);
    }
    // V: 512 chunks of 16B; chunk c -> k = c>>4, part = c&15
#pragma unroll
    for (int j = 0; j < 4; ++j) {
      int c = j * 128 + tid;
      int k = c >> 4, part = c & 15;
      async_copy16(V + (size_t)(kt2 + k) * D_ + part * 8,
                   &Vbuf[buf][k * VLD + part * 8]);
    }
  };

  // prologue: load tile kt=0
  issue_tile(0, 0);
  wait_async0();
  __syncthreads();

  int cur = 0;
  for (int kt = 0; kt < MPAD; kt += 32) {
    int nxt = cur ^ 1;
    if (kt + 32 < MPAD) issue_tile(nxt, kt + 32);   // prefetch next tile

    // S = Q @ K^T for 32 keys: two 16x16 score tiles, K-dim = 128
    v8f S0 = {}, S1 = {};
#pragma unroll
    for (int dk = 0; dk < 4; ++dk) {
      const _Float16* kp = &Kbuf[cur][(dk * 32) * KLD + r];
      v16h b0 = load_b_f16(kp, KLD, half);
      v16h b1 = load_b_f16(kp + 16, KLD, half);
      S0 = wmma16(qf[dk], b0, S0);
      S1 = wmma16(qf[dk], b1, S1);
    }
    // mask padded key (index MVALID..): only hits last column of last tile
    if (kt + 16 + r >= MVALID) {
#pragma unroll
      for (int v = 0; v < 8; ++v) S1[v] = -3.0e38f;
    }
    // online softmax; row (v + 8*half) is striped across lanes of one half
#pragma unroll
    for (int v = 0; v < 8; ++v) {
      float rm = redmax16(fmaxf(S0[v], S1[v]));
      float nm = fmaxf(mi[v], rm);
      float sc = __expf(mi[v] - nm);
      mi[v] = nm;
      float p0 = __expf(S0[v] - nm);
      float p1 = __expf(S1[v] - nm);
      li[v] = li[v] * sc + redsum16(p0 + p1);
#pragma unroll
      for (int oc = 0; oc < 8; ++oc) O[oc][v] *= sc;
      int prow = v + 8 * half;
      pstage[wave][prow * PLD + r]      = (_Float16)p0;
      pstage[wave][prow * PLD + 16 + r] = (_Float16)p1;
    }
    // C-layout -> A-layout through per-wave LDS staging (in-order DS per wave)
    __builtin_amdgcn_wave_barrier();
    asm volatile("s_wait_dscnt 0" ::: "memory");
    v16h pf;
    {
      v8h lo = *(const v8h*)&pstage[wave][r * PLD + 8 * half];
      v8h hi = *(const v8h*)&pstage[wave][r * PLD + 16 + 8 * half];
#pragma unroll
      for (int i = 0; i < 8; ++i) { pf[i] = lo[i]; pf[8 + i] = hi[i]; }
    }
    __builtin_amdgcn_wave_barrier();
    // O += P @ V  (K-dim = 32 keys, 8 column chunks of D)
#pragma unroll
    for (int oc = 0; oc < 8; ++oc) {
      const _Float16* vp = &Vbuf[cur][oc * 16 + r];
      v16h bb = load_b_f16(vp, VLD, half);
      O[oc] = wmma16(pf, bb, O[oc]);
    }

    wait_async0();        // next tile resident in LDS
    __syncthreads();      // everyone done reading cur before it is reused
    cur = nxt;
  }

  // normalize and store y (fp16)
#pragma unroll
  for (int v = 0; v < 8; ++v) {
    float inv = 1.f / li[v];
    int row = row0 + v + 8 * half;
    _Float16* yr = y + ((size_t)b * N_ + row) * D_;
#pragma unroll
    for (int oc = 0; oc < 8; ++oc)
      yr[oc * 16 + r] = (_Float16)(O[oc][v] * inv);
  }
}

// ---- kernel 4: z = x + y @ Wf --------------------------------------------
__global__ __launch_bounds__(128)
void out_kernel(const _Float16* __restrict__ y, const float* __restrict__ Wf,
                const float* __restrict__ x, float* __restrict__ z) {
  const int lane = threadIdx.x & 31, wave = threadIdx.x >> 5;
  const int r = lane & 15, half = lane >> 4;
  const int row0 = blockIdx.x * 16;
  const _Float16* Yrow = y + (size_t)row0 * D_;

  v16h af[4];
#pragma unroll
  for (int dk = 0; dk < 4; ++dk) af[dk] = load_a_f16(Yrow, D_, r, half, dk * 32);

#pragma unroll
  for (int ct = 0; ct < 4; ++ct) {
    int n0 = wave * 64 + ct * 16;
    v8f acc = {};
#pragma unroll
    for (int k0 = 0; k0 < D_; k0 += 32) {
      v16h b = load_b_f32(Wf, C_, lane, k0, n0);
      acc = wmma16(af[k0 >> 5], b, acc);
    }
#pragma unroll
    for (int v = 0; v < 8; ++v) {
      size_t idx = (size_t)(row0 + v + 8 * half) * C_ + n0 + r;
      z[idx] = x[idx] + acc[v];
    }
  }
}

// ---- launch ---------------------------------------------------------------
extern "C" void kernel_launch(void* const* d_in, const int* in_sizes, int n_in,
                              void* d_out, int out_size, void* d_ws, size_t ws_size,
                              hipStream_t stream) {
  const float* x  = (const float*)d_in[0];
  const float* Wt = (const float*)d_in[1];
  const float* Wp = (const float*)d_in[2];
  const float* Wg = (const float*)d_in[3];
  const float* Wf = (const float*)d_in[4];
  float* z = (float*)d_out;

  char* ws = (char*)d_ws;
  const size_t MB4 = (size_t)ROWS * D_ * sizeof(_Float16); // 4 MiB each
  _Float16* theta = (_Float16*)(ws + 0 * MB4);
  _Float16* phi   = (_Float16*)(ws + 1 * MB4);
  _Float16* g     = (_Float16*)(ws + 2 * MB4);
  _Float16* phiT  = (_Float16*)(ws + 3 * MB4); // [B][D][MPAD]
  _Float16* gp    = (_Float16*)(ws + 4 * MB4); // [B][MPAD][D]
  _Float16* y     = (_Float16*)(ws + 5 * MB4);

  proj_kernel<<<dim3(ROWS / 16, 6), 128, 0, stream>>>(x, Wt, Wp, Wg, theta, phi, g);
  pool_kernel<<<(B_ * MPAD * D_) / 256, 256, 0, stream>>>(phi, g, phiT, gp);
  attn_kernel<<<B_ * (N_ / 64), 128, 0, stream>>>(theta, phiT, gp, y);
  out_kernel<<<ROWS / 16, 128, 0, stream>>>(y, Wf, x, z);
}
